// VarRnnCellWrapper_88656714924167
// MI455X (gfx1250) — matrix operand order repaired and checked by reading, not compiled, not run on hardware
//
#include <hip/hip_runtime.h>
#include <hip/hip_bf16.h>

// ---------------------------------------------------------------------------
// Variational-dropout LSTM for MI455X (gfx1250), wave32 + WMMA.
//  Phase 1: x_gates = (x*mask_x) @ W_ih^T + (b_ih+b_hh)   (big f16 WMMA GEMM,
//           32x64 register tile per wave = 8 accumulators, W_ih strip in LDS)
//  Phase 2: 256 fused step kernels: gates = x_gates[s] + (h*mask_h) @ W_hh^T.
//           Wave owns 32 batch rows x 16 hidden cols x 4 gates (8 accums);
//           LSTM cell applied in-register; h-chunk LDS-staged block-wide.
//  8 accumulators/wave (64 VGPRs) chosen to stay under the VGPR budget the
//  compiler will allocate without spilling (round-2's 16-acc tile spilled).
// ---------------------------------------------------------------------------

typedef __attribute__((ext_vector_type(16))) _Float16 v16h;
typedef __attribute__((ext_vector_type(8)))  _Float16 v8h;
typedef __attribute__((ext_vector_type(8)))  float    v8f;

#define SEQ   256
#define BATCH 64
#define INDIM 1024
#define HDIM  1024
#define GDIM  4096   /* 4*H */

// ---- WMMA helpers ----------------------------------------------------------

__device__ __forceinline__ v8f wmma_f16(v16h a, v16h b, v8f c) {
    // 8 args: (neg_a, A, neg_b, B, c_mod, C, reuse_a, reuse_b)
    return __builtin_amdgcn_wmma_f32_16x16x32_f16(
        false, a, false, b, (short)0, c, false, false);
}

// A-matrix 16x32 f16 fragment from row-major X[row, k] (global).
// Lane group g: K = k0+16g..k0+16g+15 contiguous.
__device__ __forceinline__ v16h load_a_frag_g(const _Float16* __restrict__ base,
                                              int row0, int k0, int ld, int lane) {
    const int g = lane >> 4;
    const _Float16* p = base + (size_t)(row0 + (lane & 15)) * ld + k0 + g * 16;
    union { v16h v; v8h h[2]; } u;
    u.h[0] = *(const v8h*)p;
    u.h[1] = *(const v8h*)(p + 8);
    return u.v;
}

// Same A fragment but from an LDS-staged [rows][32] f16 chunk.
__device__ __forceinline__ v16h load_a_frag_lds(const _Float16* buf,
                                                int row0, int lane) {
    const int g = lane >> 4;
    const _Float16* p = buf + (row0 + (lane & 15)) * 32 + g * 16;
    union { v16h v; v8h h[2]; } u;
    u.h[0] = *(const v8h*)p;          // ds_load_b128
    u.h[1] = *(const v8h*)(p + 8);    // ds_load_b128
    return u.v;
}

// B-matrix 32x16 fragment, B[k][n] = W[col0+n][k] (computes X @ W^T), global.
// Lane group g holds K = k0+16g..k0+16g+15 contiguous: one 32B load.
__device__ __forceinline__ v16h load_b_frag_g(const _Float16* __restrict__ base,
                                              int col0, int k0, int ld, int lane) {
    const int g = lane >> 4;
    const _Float16* p = base + (size_t)(col0 + (lane & 15)) * ld + k0 + g * 16;
    return *(const v16h*)p;
}

// Same B fragment from an LDS-staged [64][32] f16 chunk (rows = output cols).
__device__ __forceinline__ v16h load_b_frag_lds(const _Float16* buf,
                                                int col0, int lane) {
    const int g = lane >> 4;
    const _Float16* p = buf + (col0 + (lane & 15)) * 32 + g * 16;
    return *(const v16h*)p;           // 2x ds_load_b128
}

__device__ __forceinline__ float sigmoidf_(float x) {
    return 1.0f / (1.0f + __expf(-x));
}

// ---- Prologue elementwise kernels -----------------------------------------

__global__ void cvt_f32_to_f16(const float* __restrict__ src,
                               _Float16* __restrict__ dst, int n) {
    int i = blockIdx.x * blockDim.x + threadIdx.x;
    if (i < n) dst[i] = (_Float16)src[i];
}

__global__ void make_bias(const float* __restrict__ b_ih,
                          const float* __restrict__ b_hh,
                          float* __restrict__ bias) {
    int i = blockIdx.x * blockDim.x + threadIdx.x;
    if (i < GDIM) bias[i] = b_ih[i] + b_hh[i];
}

__global__ void mask_cvt_x(const float* __restrict__ x,
                           const float* __restrict__ mask_x,
                           _Float16* __restrict__ xm) {
    size_t i = (size_t)blockIdx.x * blockDim.x + threadIdx.x;
    if (i >= (size_t)SEQ * BATCH * INDIM) return;
    int k = (int)(i & (INDIM - 1));
    int b = (int)((i >> 10) & (BATCH - 1));
    xm[i] = (_Float16)(x[i] * mask_x[b * INDIM + k]);
}

__global__ void init_state(const float* __restrict__ h0,
                           const float* __restrict__ c0,
                           const float* __restrict__ mask_h,
                           _Float16* __restrict__ hm,
                           float* __restrict__ c_state) {
    int i = blockIdx.x * blockDim.x + threadIdx.x;
    if (i < BATCH * HDIM) {
        hm[i] = (_Float16)(h0[i] * mask_h[i]);
        c_state[i] = c0[i];
    }
}

__global__ void finalize(const float* __restrict__ h_last,
                         const float* __restrict__ c_state,
                         float* __restrict__ out_tail) {
    int i = blockIdx.x * blockDim.x + threadIdx.x;
    if (i < BATCH * HDIM) {
        out_tail[i] = h_last[i];
        out_tail[BATCH * HDIM + i] = c_state[i];
    }
}

// ---- Phase 1: x_gates = xm @ W_ih^T + bias --------------------------------
// Block = 256 thr (8 waves stacked in M), sharing one 64-col W_ih strip
// double-buffer staged in LDS. Wave w owns a 32(M)x64(N) tile: 8 accums;
// per K-chunk: 2 A-frag global loads + 4 B-frag LDS reads -> 8 WMMAs.
__global__ void gemm_xgates(const _Float16* __restrict__ xm,   // [16384,1024]
                            const _Float16* __restrict__ Wih,  // [4096,1024]
                            const float* __restrict__ bias,    // [4096]
                            float* __restrict__ xg)            // [16384,4096]
{
    __shared__ _Float16 Bbuf[2][64 * 32];   // 2 x 4KB

    const int tid  = threadIdx.x;
    const int lane = tid & 31;
    const int wave = tid >> 5;                 // 0..7 (stacked in M)
    const int m0 = blockIdx.y * 256 + wave * 32;
    const int n0 = blockIdx.x * 64;

    // staging: thread -> 8 f16 (16B) of the 64x32 W_ih chunk
    const int srow = tid >> 2;                 // 0..63 (W_ih row n0+srow)
    const int scol = (tid & 3) * 8;            // 0,8,16,24
    const _Float16* wsrc = Wih + (size_t)(n0 + srow) * INDIM + scol;

    v8f acc[4][2] = {};                        // [n-tile][m-tile]
    int p = 0;
    for (int k0 = 0; k0 < INDIM; k0 += 32, p ^= 1) {
        *(v8h*)&Bbuf[p][tid * 8] = *(const v8h*)(wsrc + k0);
        __syncthreads();

        v16h a0 = load_a_frag_g(xm, m0,      k0, INDIM, lane);
        v16h a1 = load_a_frag_g(xm, m0 + 16, k0, INDIM, lane);
#pragma unroll
        for (int nt = 0; nt < 4; ++nt) {
            v16h b = load_b_frag_lds(Bbuf[p], nt * 16, lane);
            acc[nt][0] = wmma_f16(a0, b, acc[nt][0]);
            acc[nt][1] = wmma_f16(a1, b, acc[nt][1]);
        }
    }

    const int half = lane >> 4, n = lane & 15;
#pragma unroll
    for (int nt = 0; nt < 4; ++nt) {
        const int col = n0 + nt * 16 + n;
        const float bv = bias[col];
#pragma unroll
        for (int mt = 0; mt < 2; ++mt) {
#pragma unroll
            for (int i = 0; i < 8; ++i) {
                const int m = m0 + mt * 16 + half * 8 + i;
                xg[(size_t)m * GDIM + col] = acc[nt][mt][i] + bv;
            }
        }
    }
}

// ---- Phase 2: fused LSTM step ---------------------------------------------
// Grid = 32 blocks x 128 thr (4 waves). Wave (jt = w>>1, mh = w&1) owns
// hidden col tile j0 = blockIdx.x*32 + jt*16 for batch rows mh*32..mh*32+31
// and all 4 gates: 8 accumulators. The 64x32 h-chunk is LDS-staged and
// shared by all 4 waves; each W_hh fragment is read by only 2 waves.
__global__ void lstm_step(const _Float16* __restrict__ hm_in,  // [64,1024] f16
                          _Float16* __restrict__ hm_out,       // [64,1024] f16
                          const _Float16* __restrict__ Whh,    // [4096,1024] f16
                          const float* __restrict__ xg,        // [64,4096] (bias incl.)
                          float* __restrict__ c_state,         // [64,1024]
                          float* __restrict__ h_out,           // [64,1024] out slice
                          const float* __restrict__ mask_h)    // [64,1024]
{
    __shared__ _Float16 Abuf[2][64 * 32];   // 2 x 4KB h-chunk

    const int tid  = threadIdx.x;
    const int lane = tid & 31;
    const int wave = tid >> 5;                       // 0..3
    const int j0 = blockIdx.x * 32 + (wave >> 1) * 16;  // hidden col tile
    const int m0 = (wave & 1) * 32;                     // batch half

    // staging: thread -> 16 f16 (32B) of the 64x32 h chunk
    const int srow = tid >> 1;                       // 0..63 (batch row)
    const int scol = (tid & 1) * 16;                 // 0 or 16
    const _Float16* hsrc = hm_in + (size_t)srow * HDIM + scol;

    v8f acc[4][2] = {};                              // [gate][m-tile]
    int p = 0;
    for (int k0 = 0; k0 < HDIM; k0 += 32, p ^= 1) {
        *(v16h*)&Abuf[p][tid * 16] = *(const v16h*)(hsrc + k0);
        __syncthreads();

        if (k0 + 32 < HDIM) {
            // stream-prefetch next K-chunk of this wave's i-gate weight rows
            __builtin_prefetch(Whh + (size_t)(j0 + (lane & 15)) * HDIM + k0 + 32, 0, 1);
        }

        v16h a0 = load_a_frag_lds(Abuf[p], m0,      lane);
        v16h a1 = load_a_frag_lds(Abuf[p], m0 + 16, lane);
#pragma unroll
        for (int g = 0; g < 4; ++g) {
            v16h b = load_b_frag_g(Whh, g * HDIM + j0, k0, HDIM, lane);
            acc[g][0] = wmma_f16(a0, b, acc[g][0]);
            acc[g][1] = wmma_f16(a1, b, acc[g][1]);
        }
    }

    // LSTM cell, in-register on the accumulators.
    const int half = lane >> 4;
    const int col = j0 + (lane & 15);
#pragma unroll
    for (int mt = 0; mt < 2; ++mt) {
#pragma unroll
        for (int i = 0; i < 8; ++i) {
            const int m = m0 + mt * 16 + half * 8 + i;
            const float* xgm = xg + (size_t)m * GDIM;
            float gi = acc[0][mt][i] + xgm[col];
            float gf = acc[1][mt][i] + xgm[HDIM + col];
            float gg = acc[2][mt][i] + xgm[2 * HDIM + col];
            float go = acc[3][mt][i] + xgm[3 * HDIM + col];
            float iv = sigmoidf_(gi);
            float fv = sigmoidf_(gf);
            float gv = tanhf(gg);
            float ov = sigmoidf_(go);
            const int idx = m * HDIM + col;
            float cn = fv * c_state[idx] + iv * gv;
            float hn = ov * tanhf(cn);
            c_state[idx] = cn;
            h_out[idx] = hn;
            hm_out[idx] = (_Float16)(hn * mask_h[idx]);
        }
    }
}

// ---------------------------------------------------------------------------

extern "C" void kernel_launch(void* const* d_in, const int* in_sizes, int n_in,
                              void* d_out, int out_size, void* d_ws, size_t ws_size,
                              hipStream_t stream) {
    const float* x      = (const float*)d_in[0];
    const float* h0     = (const float*)d_in[1];
    const float* c0     = (const float*)d_in[2];
    const float* mask_x = (const float*)d_in[3];
    const float* mask_h = (const float*)d_in[4];
    const float* W_ih   = (const float*)d_in[5];
    const float* W_hh   = (const float*)d_in[6];
    const float* b_ih   = (const float*)d_in[7];
    const float* b_hh   = (const float*)d_in[8];
    float* out = (float*)d_out;

    // Workspace carve-up (~305 MB)
    char* ws = (char*)d_ws;
    _Float16* Whh_f = (_Float16*)ws;  ws += (size_t)GDIM * HDIM * sizeof(_Float16);
    _Float16* Wih_f = (_Float16*)ws;  ws += (size_t)GDIM * INDIM * sizeof(_Float16);
    float*    bias  = (float*)ws;     ws += (size_t)GDIM * sizeof(float);
    _Float16* hm0   = (_Float16*)ws;  ws += (size_t)BATCH * HDIM * sizeof(_Float16);
    _Float16* hm1   = (_Float16*)ws;  ws += (size_t)BATCH * HDIM * sizeof(_Float16);
    float* c_state  = (float*)ws;     ws += (size_t)BATCH * HDIM * sizeof(float);
    _Float16* xm    = (_Float16*)ws;  ws += (size_t)SEQ * BATCH * INDIM * sizeof(_Float16);
    float*    xg    = (float*)ws;     // [16384, 4096] f32

    // Prologue
    {
        const int n = GDIM * HDIM;
        cvt_f32_to_f16<<<(n + 255) / 256, 256, 0, stream>>>(W_hh, Whh_f, n);
        cvt_f32_to_f16<<<(n + 255) / 256, 256, 0, stream>>>(W_ih, Wih_f, n);
        make_bias<<<(GDIM + 255) / 256, 256, 0, stream>>>(b_ih, b_hh, bias);
        const size_t nx = (size_t)SEQ * BATCH * INDIM;
        mask_cvt_x<<<(unsigned)((nx + 255) / 256), 256, 0, stream>>>(x, mask_x, xm);
        init_state<<<(BATCH * HDIM + 255) / 256, 256, 0, stream>>>(h0, c0, mask_h, hm0, c_state);
    }

    // Phase 1: input-projection GEMM (16384 x 4096 x 1024)
    {
        dim3 grid(GDIM / 64, (SEQ * BATCH) / 256);
        gemm_xgates<<<grid, 256, 0, stream>>>(xm, Wih_f, bias, xg);
    }

    // Phase 2: 256 fused recurrent steps (ping-pong masked-h buffers)
    for (int s = 0; s < SEQ; ++s) {
        const _Float16* hin  = (s & 1) ? hm1 : hm0;
        _Float16*       hout = (s & 1) ? hm0 : hm1;
        lstm_step<<<HDIM / 32, 128, 0, stream>>>(
            hin, hout, Whh_f,
            xg + (size_t)s * BATCH * GDIM,
            c_state,
            out + (size_t)s * BATCH * HDIM,
            mask_h);
    }

    // Epilogue: h_n (copy of last slice) and c_n
    finalize<<<(BATCH * HDIM + 255) / 256, 256, 0, stream>>>(
        out + (size_t)(SEQ - 1) * BATCH * HDIM, c_state,
        out + (size_t)SEQ * BATCH * HDIM);
}
